// DetectorNetwork_87686052315388
// MI455X (gfx1250) — compile-verified, matrix operand order checked by reading, not run on hardware
//
#include <hip/hip_runtime.h>
#include <math.h>

// ---------------------------------------------------------------------------
// DetectorNetwork head for MI455X (gfx1250, wave32, WMMA).
// - clean+noise batches fused into one M=1024 fc6/fc7 GEMM (fc6_w streamed once)
// - all weights/activations pre-converted to bf16: zero conversion VALU in the
//   WMMA hot loops
// - staging via GLOBAL_LOAD_ASYNC_TO_LDS_B128 (ASYNCcnt); template-specialized
//   full-tile path is branch-free (no exec-mask ops in the fc6/fc7 inner loop)
// ---------------------------------------------------------------------------

typedef __attribute__((ext_vector_type(16))) __bf16 v16bf;
typedef __attribute__((ext_vector_type(8)))  float  v8f;
typedef unsigned short u16;

#define ROIS   512
#define CCH    512
#define FH     50
#define FW     50
#define OUTP   7
#define FLATSZ 25088   // 512*7*7
#define FVS    4096
#define NCLS   21
#define NREG   80
#define GLSTR  20      // LDS row stride in uints (16 K-pairs + 4 pad, 16B-aligned)

// ---- async global->LDS availability (guarded; safe fallback) --------------
#if defined(__has_builtin)
# if __has_builtin(__builtin_amdgcn_global_load_async_to_lds_b128) && \
     __has_builtin(__builtin_amdgcn_s_wait_asynccnt)
#  define USE_ASYNC_LDS 1
# endif
#endif
#ifndef USE_ASYNC_LDS
# define USE_ASYNC_LDS 0
#endif

#if USE_ASYNC_LDS
// builtin signature (from hipcc diagnostic): param0 = v4i global*, param1 = v4i lds*
typedef int v4i_b __attribute__((vector_size(16)));
typedef __attribute__((address_space(1))) v4i_b* gv4i_p;   // global
typedef __attribute__((address_space(3))) v4i_b* lv4i_p;   // LDS
__device__ __forceinline__ void async_cp16(const void* g, void* l) {
  __builtin_amdgcn_global_load_async_to_lds_b128((gv4i_p)g, (lv4i_p)l, 0, 0);
}
__device__ __forceinline__ void async_wait0() {
  __builtin_amdgcn_s_wait_asynccnt(0);
}
#else
__device__ __forceinline__ void async_cp16(const void* g, void* l) {
  *(uint4*)l = *(const uint4*)g;
}
__device__ __forceinline__ void async_wait0() {}
#endif

// ---- bf16 helpers (round-to-nearest-even) ---------------------------------
__device__ __forceinline__ u16 f2bf(float x) {
  union { float f; unsigned u; } a; a.f = x;
  unsigned u = a.u + (0x7FFFu + ((a.u >> 16) & 1u));
  return (u16)(u >> 16);
}
__device__ __forceinline__ float bf2f(u16 h) {
  union { unsigned u; float f; } a; a.u = ((unsigned)h) << 16;
  return a.f;
}
__device__ __forceinline__ unsigned pack_bf16x2(float lo, float hi) {
  return (unsigned)f2bf(lo) | ((unsigned)f2bf(hi) << 16);
}
// fragment K-pair index for VGPR v given lane's ksel (lane>>4):
// lanes 0-15 hold K pairs {0..3, 8..11}, lanes 16-31 hold {4..7, 12..15}
__device__ __forceinline__ int frag_pair(int v, int ksel) {
  return (v < 4 ? 0 : 8) + ksel * 4 + (v & 3);
}

// ---------------------------------------------------------------------------
// 0) fp32 -> bf16 bulk conversion (memory-bound; VALU cost hidden by HBM)
// ---------------------------------------------------------------------------
__global__ void cvt_f32_bf16(const float* __restrict__ src, u16* __restrict__ dst,
                             size_t n4) {
  size_t i = (size_t)blockIdx.x * blockDim.x + threadIdx.x;
  size_t stride = (size_t)gridDim.x * blockDim.x;
  for (; i < n4; i += stride) {
    float4 v = ((const float4*)src)[i];
    uint2 o;
    o.x = pack_bf16x2(v.x, v.y);
    o.y = pack_bf16x2(v.z, v.w);
    ((uint2*)dst)[i] = o;
  }
}

// ---------------------------------------------------------------------------
// 1) ROI max pooling -> bf16 rows 0..511 of X
// ---------------------------------------------------------------------------
__global__ void roi_pool_kernel(const float* __restrict__ fmap,   // (512,50,50)
                                const float* __restrict__ prop,   // (512,4) y1,x1,y2,x2
                                u16* __restrict__ Xbf)            // (1024,25088)
{
  int roi = blockIdx.x;
  float y1 = prop[roi * 4 + 0], x1 = prop[roi * 4 + 1];
  float y2 = prop[roi * 4 + 2], x2 = prop[roi * 4 + 3];
  const float S = 1.0f / 16.0f;
  float rsw = rintf(x1 * S), rsh = rintf(y1 * S);
  float rew = rintf(x2 * S), reh = rintf(y2 * S);
  float bw = fmaxf(rew - rsw + 1.0f, 1.0f) / 7.0f;
  float bh = fmaxf(reh - rsh + 1.0f, 1.0f) / 7.0f;

  for (int c = threadIdx.x; c < CCH; c += blockDim.x) {
    const float* fm = fmap + (size_t)c * (FH * FW);
    u16* dst = Xbf + (size_t)roi * FLATSZ + (size_t)c * (OUTP * OUTP);
    for (int ph = 0; ph < OUTP; ++ph) {
      float hs = fminf(fmaxf(floorf((float)ph * bh) + rsh, 0.0f), 50.0f);
      float he = fminf(fmaxf(ceilf((float)(ph + 1) * bh) + rsh, 0.0f), 50.0f);
      int hi = (int)hs;
      for (int pw = 0; pw < OUTP; ++pw) {
        float wss = fminf(fmaxf(floorf((float)pw * bw) + rsw, 0.0f), 50.0f);
        float wee = fminf(fmaxf(ceilf((float)(pw + 1) * bw) + rsw, 0.0f), 50.0f);
        int wi = (int)wss;
        bool empty = (he <= hs) || (wee <= wss);
        float m = -INFINITY;
        for (int a = 0; a < 10; ++a) {
          int ih = hi + a;
          if ((float)ih >= he) break;
          for (int b = 0; b < 10; ++b) {
            int iw = wi + b;
            if ((float)iw >= wee) break;
            m = fmaxf(m, fm[ih * FW + iw]);
          }
        }
        dst[ph * OUTP + pw] = f2bf(empty ? 0.0f : m);
      }
    }
  }
}

// ---------------------------------------------------------------------------
// 2) Generic WMMA GEMM on bf16:  C(M,N) = act(A(M,K) . W(N,K)^T + bias)
//    M multiple of 128, K multiple of 32. FULLN=true -> branch-free staging
//    (all N tiles full); FULLN=false -> N-guarded edge path.
// ---------------------------------------------------------------------------
template <bool FULLN>
__global__ __launch_bounds__(256)
void gemm_bf16(const u16* __restrict__ A, const u16* __restrict__ W,
               const float* __restrict__ bias, void* __restrict__ Cout,
               int M, int N, int K, int act, int outBf)
{
  __shared__ unsigned As[128 * GLSTR];
  __shared__ unsigned Bs[128 * GLSTR];

  const int tid   = threadIdx.x;
  const int lane  = tid & 31;
  const int wave  = tid >> 5;     // 0..7
  const int waveM = wave >> 2;    // 0..1 -> 64 rows
  const int waveN = wave & 3;     // 0..3 -> 32 cols
  const int m0 = blockIdx.x * 128;
  const int n0 = blockIdx.y * 128;
  const int lrow = lane & 15;
  const int ksel = lane >> 4;

  v8f acc[4][2];
#pragma unroll
  for (int i = 0; i < 4; ++i)
#pragma unroll
    for (int j = 0; j < 2; ++j) acc[i][j] = (v8f){0,0,0,0,0,0,0,0};

  for (int kt = 0; kt < K; kt += 32) {
    // stage A and B tiles: 128 rows x 16 uints (= 32 bf16), 4 uint4 per row
#pragma unroll
    for (int i = 0; i < 2; ++i) {
      int gid = tid + i * 256;          // 0..511
      int row = gid >> 2;               // 0..127
      int ch  = gid & 3;                // uint4 chunk (8 bf16)
      const u16* ag = A + (size_t)(m0 + row) * K + kt + ch * 8;
      const u16* bg = W + (size_t)(n0 + row) * K + kt + ch * 8;
      unsigned* al = &As[row * GLSTR + ch * 4];
      unsigned* bl = &Bs[row * GLSTR + ch * 4];
      async_cp16(ag, al);
      if (FULLN || (n0 + row) < N) async_cp16(bg, bl);
      else                         *(uint4*)bl = make_uint4(0u, 0u, 0u, 0u);
      // prefetch next K tile (global_prefetch_b8; speculative, OOB-safe)
      int ktn = (kt + 32 < K) ? (kt + 32) : kt;
      __builtin_prefetch(A + (size_t)(m0 + row) * K + ktn + ch * 8, 0, 1);
      __builtin_prefetch(W + (size_t)(n0 + row) * K + ktn + ch * 8, 0, 1);
    }
    async_wait0();
    __syncthreads();

    union Frag { v16bf v; unsigned u[8]; };
    Frag bf[2];
#pragma unroll
    for (int nf = 0; nf < 2; ++nf) {
      int nb = waveN * 32 + nf * 16 + lrow;
#pragma unroll
      for (int v = 0; v < 8; ++v) bf[nf].u[v] = Bs[nb * GLSTR + frag_pair(v, ksel)];
    }
#pragma unroll
    for (int mf = 0; mf < 4; ++mf) {
      Frag af;
      int mb = waveM * 64 + mf * 16 + lrow;
#pragma unroll
      for (int v = 0; v < 8; ++v) af.u[v] = As[mb * GLSTR + frag_pair(v, ksel)];
#pragma unroll
      for (int nf = 0; nf < 2; ++nf) {
        acc[mf][nf] = __builtin_amdgcn_wmma_f32_16x16x32_bf16(
            false, af.v, false, bf[nf].v, (short)0, acc[mf][nf], false, false);
      }
    }
    __syncthreads();
  }

#pragma unroll
  for (int nf = 0; nf < 2; ++nf) {
    int col = n0 + waveN * 32 + nf * 16 + lrow;
    if (!FULLN && col >= N) continue;
    float b = bias[col];
#pragma unroll
    for (int mf = 0; mf < 4; ++mf) {
      int rbase = m0 + waveM * 64 + mf * 16 + ksel * 8;
#pragma unroll
      for (int r = 0; r < 8; ++r) {
        float v = acc[mf][nf][r] + b;
        if (act == 1) v = fmaxf(v, 0.0f);
        if (outBf) ((u16*)Cout)[(size_t)(rbase + r) * N + col] = f2bf(v);
        else       ((float*)Cout)[(size_t)(rbase + r) * N + col] = v;
      }
    }
  }
}

// ---------------------------------------------------------------------------
// 3) 4x4 VALID conv as implicit-im2col WMMA GEMM (bf16 in/out).
//    M=16 positions, K=8192 (c*16+kh*4+kw), Out[roi][o*16+pos] so the
//    (N,16,512) reshape rows are contiguous 512-element chunks.
// ---------------------------------------------------------------------------
__global__ __launch_bounds__(256)
void conv_roi_wmma(const u16* __restrict__ Xbf,     // clean rois, rows 0..511
                   const u16* __restrict__ Wc,      // (512, 8192) bf16
                   const float* __restrict__ bc,    // (512,)
                   u16* __restrict__ Out)           // (512, 8192) bf16
{
  __shared__ __align__(16) u16 inp[CCH * 49];       // bf16 ROI tile (~50KB)
  __shared__ unsigned Asm[16 * GLSTR];
  __shared__ unsigned Bsm[128 * GLSTR];

  const int roi = blockIdx.y;
  const int o0  = blockIdx.x * 128;
  const int tid = threadIdx.x;
  const int lane = tid & 31;
  const int wave = tid >> 5;
  const int lrow = lane & 15;
  const int ksel = lane >> 4;

  // copy ROI tile (already bf16): 25088 u16 = 3136 uint4
  {
    const uint4* src = (const uint4*)(Xbf + (size_t)roi * FLATSZ);
    uint4* dst = (uint4*)inp;
    for (int i = tid; i < 3136; i += 256) async_cp16(src + i, dst + i);
  }
  async_wait0();
  __syncthreads();

  v8f acc = (v8f){0,0,0,0,0,0,0,0};

  for (int kt = 0; kt < 8192; kt += 32) {
    // stage weights: 128 channels x 16 uints
#pragma unroll
    for (int i = 0; i < 2; ++i) {
      int gid = tid + i * 256;
      int row = gid >> 2, ch = gid & 3;
      const u16* bg = Wc + (size_t)(o0 + row) * 8192 + kt + ch * 8;
      async_cp16(bg, &Bsm[row * GLSTR + ch * 4]);
    }
    // stage im2col A (16 positions x 16 K-pairs), one pair per thread
    {
      int pos = tid >> 4;        // 0..15
      int pr  = tid & 15;        // 0..15
      int oh = pos >> 2, ow = pos & 3;
      u16 h[2];
#pragma unroll
      for (int s = 0; s < 2; ++s) {
        int k = kt + pr * 2 + s;
        int c = k >> 4, rem = k & 15, kh = rem >> 2, kw = rem & 3;
        h[s] = inp[c * 49 + (oh + kh) * 7 + (ow + kw)];
      }
      Asm[pos * GLSTR + pr] = (unsigned)h[0] | ((unsigned)h[1] << 16);
    }
    async_wait0();
    __syncthreads();

    union Frag { v16bf v; unsigned u[8]; };
    Frag af, bfr;
#pragma unroll
    for (int v = 0; v < 8; ++v) {
      int p = frag_pair(v, ksel);
      af.u[v]  = Asm[lrow * GLSTR + p];
      bfr.u[v] = Bsm[(wave * 16 + lrow) * GLSTR + p];
    }
    acc = __builtin_amdgcn_wmma_f32_16x16x32_bf16(
        false, af.v, false, bfr.v, (short)0, acc, false, false);
    __syncthreads();
  }

  int o = o0 + wave * 16 + lrow;
  float b = bc[o];
#pragma unroll
  for (int r = 0; r < 8; ++r) {
    int pos = ksel * 8 + r;
    float v = fmaxf(acc[r] + b, 0.0f);                // relu
    Out[(size_t)roi * 8192 + o * 16 + pos] = f2bf(v);
  }
}

// ---------------------------------------------------------------------------
// 4) ns = mean(sigmoid(fc2)) over 16 rows per ROI
// ---------------------------------------------------------------------------
__global__ void ns_finalize_kernel(const float* __restrict__ Sv, float* __restrict__ ns) {
  int r = blockIdx.x * blockDim.x + threadIdx.x;
  if (r < ROIS) {
    float s = 0.0f;
    for (int i = 0; i < 16; ++i) s += 1.0f / (1.0f + expf(-Sv[r * 16 + i]));
    ns[r] = s * (1.0f / 16.0f);
  }
}

// ---------------------------------------------------------------------------
// 5) X[512+r] = X[r] + ns[r] * sqrt(-2 ln u)   (bf16 in/out)
// ---------------------------------------------------------------------------
__global__ void add_noise_kernel(u16* __restrict__ Xbf, const float* __restrict__ u,
                                 const float* __restrict__ ns) {
  const size_t total = (size_t)ROIS * FLATSZ;
  size_t i = (size_t)blockIdx.x * blockDim.x + threadIdx.x;
  size_t stride = (size_t)gridDim.x * blockDim.x;
  for (; i < total; i += stride) {
    size_t r = i / FLATSZ;
    float n = ns[r] * sqrtf(-2.0f * logf(u[i]));
    Xbf[total + i] = f2bf(bf2f(Xbf[i]) + n);
  }
}

// ---------------------------------------------------------------------------
// 6) softmax(cls) + scatter into d_out:
//    [classes(512,21) | box(512,80) | classes_noise(512,21) | box_noise(512,80)]
// ---------------------------------------------------------------------------
__global__ void softmax_scatter_kernel(const float* __restrict__ Lcls,
                                       const float* __restrict__ Lreg,
                                       float* __restrict__ out) {
  __shared__ float sv[NCLS];
  __shared__ float ssum;
  int row = blockIdx.x;   // 0..1023
  int t = threadIdx.x;
  if (t < NCLS) sv[t] = Lcls[row * NCLS + t];
  __syncthreads();
  if (t == 0) {
    float m = sv[0];
    for (int i = 1; i < NCLS; ++i) m = fmaxf(m, sv[i]);
    float s = 0.0f;
    for (int i = 0; i < NCLS; ++i) { sv[i] = expf(sv[i] - m); s += sv[i]; }
    ssum = s;
  }
  __syncthreads();
  const size_t blk = (size_t)ROIS * NCLS + (size_t)ROIS * NREG;
  size_t clsBase = (row < ROIS) ? (size_t)row * NCLS
                                : blk + (size_t)(row - ROIS) * NCLS;
  size_t regBase = (row < ROIS) ? (size_t)ROIS * NCLS + (size_t)row * NREG
                                : blk + (size_t)ROIS * NCLS + (size_t)(row - ROIS) * NREG;
  if (t < NCLS) out[clsBase + t] = sv[t] / ssum;
  for (int j = t; j < NREG; j += blockDim.x) out[regBase + j] = Lreg[row * NREG + j];
}

// ---------------------------------------------------------------------------
// launch
// ---------------------------------------------------------------------------
extern "C" void kernel_launch(void* const* d_in, const int* in_sizes, int n_in,
                              void* d_out, int out_size, void* d_ws, size_t ws_size,
                              hipStream_t stream) {
  const float* fmap      = (const float*)d_in[0];
  const float* proposals = (const float*)d_in[1];
  const float* noise_u   = (const float*)d_in[2];
  const float* np_conv_w = (const float*)d_in[3];
  const float* np_conv_b = (const float*)d_in[4];
  const float* np_fc1_w  = (const float*)d_in[5];
  const float* np_fc1_b  = (const float*)d_in[6];
  const float* np_fc2_w  = (const float*)d_in[7];
  const float* np_fc2_b  = (const float*)d_in[8];
  const float* fc6_w     = (const float*)d_in[9];
  const float* fc6_b     = (const float*)d_in[10];
  const float* fc7_w     = (const float*)d_in[11];
  const float* fc7_b     = (const float*)d_in[12];
  const float* cls_w     = (const float*)d_in[13];
  const float* cls_b     = (const float*)d_in[14];
  const float* reg_w     = (const float*)d_in[15];
  const float* reg_b     = (const float*)d_in[16];

  // ---- workspace carve-out (256B aligned blocks) ----
  char* p = (char*)d_ws;
  auto alloc = [&](size_t bytes) -> void* {
    void* r = (void*)p; p += (bytes + 255) & ~(size_t)255; return r;
  };
  u16* W6bf  = (u16*)alloc((size_t)FVS * FLATSZ * 2);
  u16* W7bf  = (u16*)alloc((size_t)FVS * FVS * 2);
  u16* Wcbf  = (u16*)alloc((size_t)512 * 8192 * 2);
  u16* W1bf  = (u16*)alloc((size_t)512 * 512 * 2);
  u16* W2bf  = (u16*)alloc((size_t)512 * 2);
  u16* Wclsb = (u16*)alloc((size_t)NCLS * FVS * 2);
  u16* Wregb = (u16*)alloc((size_t)NREG * FVS * 2);
  u16* Xbf   = (u16*)alloc((size_t)1024 * FLATSZ * 2);
  u16* Cvbf  = (u16*)alloc((size_t)8192 * 512 * 2);   // ConvOut, aliased as Y6
  u16* Hnbf  = (u16*)alloc((size_t)8192 * 512 * 2);   // Hnp, aliased as Y7
  float* Sv  = (float*)alloc((size_t)8192 * 4);
  float* nsv = (float*)alloc((size_t)512 * 4);
  float* Lc  = (float*)alloc((size_t)1024 * NCLS * 4);
  float* Lr  = (float*)alloc((size_t)1024 * NREG * 4);
  u16* Y6bf = Cvbf;   // ConvOut dead after np_fc1
  u16* Y7bf = Hnbf;   // Hnp dead after np_fc2

  auto cvt = [&](const float* s, u16* d, size_t n) {
    size_t n4 = n / 4;
    int blocks = (int)((n4 + 255) / 256);
    if (blocks > 4096) blocks = 4096;
    cvt_f32_bf16<<<blocks, 256, 0, stream>>>(s, d, n4);
  };

  // ---- 0) one-time (per launch) weight conversion to bf16 ----
  cvt(fc6_w,     W6bf,  (size_t)FVS * FLATSZ);
  cvt(fc7_w,     W7bf,  (size_t)FVS * FVS);
  cvt(np_conv_w, Wcbf,  (size_t)512 * 8192);
  cvt(np_fc1_w,  W1bf,  (size_t)512 * 512);
  cvt(np_fc2_w,  W2bf,  (size_t)512);
  cvt(cls_w,     Wclsb, (size_t)NCLS * FVS);
  cvt(reg_w,     Wregb, (size_t)NREG * FVS);

  // ---- 1) ROI pool -> clean bf16 rows of X ----
  roi_pool_kernel<<<ROIS, 256, 0, stream>>>(fmap, proposals, Xbf);

  // ---- 2) noise-scale conv (im2col WMMA) ----
  conv_roi_wmma<<<dim3(4, ROIS), 256, 0, stream>>>(Xbf, Wcbf, np_conv_b, Cvbf);

  // ---- 3) np_fc1 relu: (8192,512)x(512,512)^T ----
  gemm_bf16<true><<<dim3(64, 4), 256, 0, stream>>>(Cvbf, W1bf, np_fc1_b, Hnbf,
                                                   8192, 512, 512, 1, 1);
  // ---- 4) np_fc2: (8192,512)x(1,512)^T -> fp32 ----
  gemm_bf16<false><<<dim3(64, 1), 256, 0, stream>>>(Hnbf, W2bf, np_fc2_b, Sv,
                                                    8192, 1, 512, 0, 0);
  // ---- 5) sigmoid + mean -> ns ----
  ns_finalize_kernel<<<2, 256, 0, stream>>>(Sv, nsv);

  // ---- 6) noise rows of X ----
  add_noise_kernel<<<2048, 256, 0, stream>>>(Xbf, noise_u, nsv);

  // ---- 7) fc6 fused clean+noise: (1024,25088)x(4096,25088)^T relu ----
  gemm_bf16<true><<<dim3(8, 32), 256, 0, stream>>>(Xbf, W6bf, fc6_b, Y6bf,
                                                   1024, FVS, FLATSZ, 1, 1);
  // ---- 8) fc7 relu ----
  gemm_bf16<true><<<dim3(8, 32), 256, 0, stream>>>(Y6bf, W7bf, fc7_b, Y7bf,
                                                   1024, FVS, FVS, 1, 1);
  // ---- 9) heads -> fp32 logits ----
  gemm_bf16<false><<<dim3(8, 1), 256, 0, stream>>>(Y7bf, Wclsb, cls_b, Lc,
                                                   1024, NCLS, FVS, 0, 0);
  gemm_bf16<false><<<dim3(8, 1), 256, 0, stream>>>(Y7bf, Wregb, reg_b, Lr,
                                                   1024, NREG, FVS, 0, 0);

  // ---- 10) softmax + scatter ----
  softmax_scatter_kernel<<<1024, 128, 0, stream>>>(Lc, Lr, (float*)d_out);

  (void)in_sizes; (void)n_in; (void)out_size; (void)ws_size;
}
